// MainModel_57750130262406
// MI455X (gfx1250) — matrix operand order repaired and checked by reading, not compiled, not run on hardware
//
#include <hip/hip_runtime.h>
#include <hip/hip_bf16.h>

typedef unsigned short u16;
typedef unsigned int u32;
typedef __attribute__((ext_vector_type(16))) __bf16 v16bf;
typedef __attribute__((ext_vector_type(8)))  float  v8f;
typedef __attribute__((ext_vector_type(4)))  unsigned v4u;
typedef __attribute__((ext_vector_type(8)))  int   v8i;
typedef __attribute__((ext_vector_type(4)))  int   v4i;

#define NT 2048          // tokens per batch
#define BATCH 8
#define NHEAD 4

#if defined(__has_builtin)
#if __has_builtin(__builtin_amdgcn_update_dpp)
#define HAVE_DPP 1
#endif
#if __has_builtin(__builtin_amdgcn_tensor_load_to_lds) && \
    __has_builtin(__builtin_amdgcn_s_wait_tensorcnt)
#define HAVE_TDM 1
#endif
#endif

union Frag { v16bf v; u16 u[16]; uint4 q[2]; };

__device__ __forceinline__ u16 f2bf(float f) {
  unsigned u = __float_as_uint(f);
  return (u16)((u + 0x7FFFu + ((u >> 16) & 1u)) >> 16);
}

template <bool RA>
__device__ __forceinline__ v8f wmma_bf16(const Frag& a, const Frag& b, v8f c) {
  return __builtin_amdgcn_wmma_f32_16x16x32_bf16(false, a.v, false, b.v,
                                                 (short)0, c, RA, false);
}

// ---------------------------------------------------------------------------
// 16-lane butterfly reductions via DPP16 (no LDS, no dscnt waits)
// quad_perm(1,0,3,2)=0xB1, quad_perm(2,3,0,1)=0x4E,
// row_half_mirror=0x141, row_mirror=0x140
// ---------------------------------------------------------------------------
#if HAVE_DPP
template <int CTRL>
__device__ __forceinline__ float dpp_mov(float x) {
  return __int_as_float(
      __builtin_amdgcn_update_dpp(0, __float_as_int(x), CTRL, 0xf, 0xf, true));
}
__device__ __forceinline__ float max16(float x) {
  x = fmaxf(x, dpp_mov<0xB1>(x));
  x = fmaxf(x, dpp_mov<0x4E>(x));
  x = fmaxf(x, dpp_mov<0x141>(x));
  x = fmaxf(x, dpp_mov<0x140>(x));
  return x;
}
__device__ __forceinline__ float sum16(float x) {
  x += dpp_mov<0xB1>(x);
  x += dpp_mov<0x4E>(x);
  x += dpp_mov<0x141>(x);
  x += dpp_mov<0x140>(x);
  return x;
}
#else
__device__ __forceinline__ float max16(float x) {
  for (int off = 1; off < 16; off <<= 1) x = fmaxf(x, __shfl_xor(x, off, 32));
  return x;
}
__device__ __forceinline__ float sum16(float x) {
  for (int off = 1; off < 16; off <<= 1) x += __shfl_xor(x, off, 32);
  return x;
}
#endif

// ---------------------------------------------------------------------------
// Tensor Data Mover: 2D bf16 tile (rows x cols) from row-major global (ld)
// into LDS. D# groups packed per CDNA5 ISA 8.3/8.4.
// ---------------------------------------------------------------------------
#if HAVE_TDM
__device__ __forceinline__ void tdm_load_2d(const void* gaddr, unsigned lds_off,
                                            int rows, int cols, int ld) {
  unsigned long long ga = (unsigned long long)gaddr;
  v4u g0;
  g0[0] = 1u;                                   // count=1, user mode
  g0[1] = lds_off;                              // lds_addr (bytes)
  g0[2] = (u32)ga;                              // global_addr[31:0]
  g0[3] = (u32)((ga >> 32) & 0x1FFFFFFu) | (2u << 30);  // addr[56:32], type=2
  v8i g1;
  g1[0] = (1 << 16);                            // data_size = 2 bytes
  g1[1] = (int)(((u32)ld & 0xFFFFu) << 16);     // tensor_dim0[15:0]
  g1[2] = (int)(((u32)ld >> 16) | (((u32)rows & 0xFFFFu) << 16));
  g1[3] = (int)(((u32)rows >> 16) | ((u32)cols << 16));  // tile_dim0
  g1[4] = rows;                                 // tile_dim1 (tile_dim2=0)
  g1[5] = ld;                                   // tensor_dim0_stride[31:0]
  g1[6] = 0;
  g1[7] = 0;
  v4i z4 = {0, 0, 0, 0};
#if __clang_major__ >= 23
  v8i z8 = {0, 0, 0, 0, 0, 0, 0, 0};
  __builtin_amdgcn_tensor_load_to_lds(g0, g1, z4, z4, z8, 0);
#else
  __builtin_amdgcn_tensor_load_to_lds(g0, g1, z4, z4, 0);
#endif
}
#endif

// ---------------------------------------------------------------------------
// fp32 -> bf16 conversion kernels
// ---------------------------------------------------------------------------
__global__ __launch_bounds__(256) void cvt_f32_bf16(const float* __restrict__ x,
                                                    u16* __restrict__ y, int n) {
  int i = blockIdx.x * 256 + threadIdx.x;
  if (i < n) y[i] = f2bf(x[i]);
}

__global__ __launch_bounds__(256) void cvt_act_kernel(const float* __restrict__ act,
                                                      u16* __restrict__ catb) {
  int i = blockIdx.x * 256 + threadIdx.x;          // total 8*524288
  int b = i >> 19;
  int r = i & ((1 << 19) - 1);
  catb[(long long)b * (512LL * NT) + r] = f2bf(act[i]);
}

__global__ __launch_bounds__(256) void zero_kernel(float* __restrict__ p, int n) {
  int i = blockIdx.x * 256 + threadIdx.x;
  if (i < n) p[i] = 0.f;
}

// ---------------------------------------------------------------------------
// WMMA GEMM:  Y(b) = A[MxK] * X(b)[K x NT] + bias
// MODE 0: store bf16       MODE 1: relu + store bf16
// MODE 2: fp32 accumulate  MODE 3: relu + global max-pool over tokens
// ---------------------------------------------------------------------------
template <int MODE>
__global__ __launch_bounds__(256) void gemm_bf16(
    const u16* __restrict__ A, const u16* __restrict__ X,
    const float* __restrict__ bias, u16* __restrict__ Yb, float* __restrict__ Yf,
    int M, int K, long long xStride, long long yStride) {
  __shared__ u16 sA[2][64 * 32];    // [m][k] row-major, k contiguous
  __shared__ u16 sX[2][128 * 32];   // [n][k] transposed, k contiguous

  const int tid = threadIdx.x;
  const int b = blockIdx.z;
  const int mBase = blockIdx.y * 64;
  const int nBase = blockIdx.x * 128;
  const u16* Xb = X + (long long)b * xStride;
  const int w = tid >> 5, lane = tid & 31;

  auto stage = [&](int buf, int kk) {
#if HAVE_TDM
    // Wave 0 DMAs the 64x32 A tile into LDS (EXEC-independent, one issue).
    if (w == 0) {
      tdm_load_2d(&A[(long long)mBase * K + kk],
                  (unsigned)(unsigned long long)(void*)&sA[buf][0], 64, 32, K);
    }
#else
    int m = tid >> 2, kg = (tid & 3) * 8;
    *(uint4*)&sA[buf][m * 32 + kg] =
        *(const uint4*)&A[(long long)(mBase + m) * K + kk + kg];
#endif
    // X tile 32x128, transpose into [n][k]
    int k = tid >> 3, n0 = (tid & 7) * 16;
    const u16* xp = &Xb[(long long)(kk + k) * NT + nBase + n0];
    uint4 x0 = *(const uint4*)xp;
    uint4 x1 = *(const uint4*)(xp + 8);
    u16 tmp[16];
    *(uint4*)&tmp[0] = x0;
    *(uint4*)&tmp[8] = x1;
#pragma unroll
    for (int e = 0; e < 16; ++e) sX[buf][(n0 + e) * 32 + k] = tmp[e];
    if (kk + 64 < K) __builtin_prefetch(xp + 64LL * NT, 0, 0);
  };

  const int am = lane & 15, ah = lane >> 4;
  const int mt = (w & 3) * 16;        // wave's M sub-tile
  const int ngb = (w >> 2) * 64;      // wave's N group base

  v8f zero = {0.f, 0.f, 0.f, 0.f, 0.f, 0.f, 0.f, 0.f};
  v8f acc[4] = {zero, zero, zero, zero};

  stage(0, 0);
  int buf = 0;
  for (int kk = 0; kk < K; kk += 32) {
#if HAVE_TDM
    if (w == 0) __builtin_amdgcn_s_wait_tensorcnt((short)0);
#endif
    __syncthreads();
    if (kk + 32 < K) stage(buf ^ 1, kk + 32);

    Frag a;
    const u16* ap = &sA[buf][(mt + am) * 32 + ah * 8];
    a.q[0] = *(const uint4*)ap;          // k: ah*8 .. +7
    a.q[1] = *(const uint4*)(ap + 16);   // k: ah*8+16 .. +7
#pragma unroll
    for (int j = 0; j < 4; ++j) {
      Frag bb;
      const u16* bp = &sX[buf][(ngb + j * 16 + am) * 32 + ah * 8];
      bb.q[0] = *(const uint4*)bp;       // k: ah*8 .. +7
      bb.q[1] = *(const uint4*)(bp + 16); // k: ah*8+16 .. +7
      if (j == 0)
        acc[j] = wmma_bf16<false>(a, bb, acc[j]);
      else
        acc[j] = wmma_bf16<true>(a, bb, acc[j]);  // same A as previous wmma
    }
    buf ^= 1;
  }

  const long long yoffb = (long long)b * yStride;
#pragma unroll
  for (int j = 0; j < 4; ++j) {
    int col = nBase + ngb + j * 16 + am;
#pragma unroll
    for (int r = 0; r < 8; ++r) {
      int row = mBase + mt + r + 8 * ah;
      float v = acc[j][r] + bias[row];
      if (MODE == 0) {
        Yb[yoffb + (long long)row * NT + col] = f2bf(v);
      } else if (MODE == 1) {
        Yb[yoffb + (long long)row * NT + col] = f2bf(fmaxf(v, 0.f));
      } else if (MODE == 2) {
        Yf[yoffb + (long long)row * NT + col] += v;
      } else {
        float mv = max16(fmaxf(v, 0.f));
        if (am == 0)
          atomicMax((unsigned int*)&Yf[(long long)b * 2048 + row],
                    __float_as_uint(mv));
      }
    }
  }
}

// ---------------------------------------------------------------------------
// Flash attention per (b, head). Wave handles 16 queries; streams 32-key
// blocks through LDS. Reference scales by 1/sqrt(HD) AFTER softmax.
// ---------------------------------------------------------------------------
__global__ __launch_bounds__(256) void attn_kernel(const u16* __restrict__ Q,
                                                   const u16* __restrict__ Km,
                                                   const u16* __restrict__ Vm,
                                                   u16* __restrict__ Mo) {
  __shared__ u16 kt[32 * 64];       // [key][d]
  __shared__ u16 vl[64 * 32];       // [d][key]
  __shared__ u16 pls[8][16 * 32];   // per-wave P staging [q][key]

  const int tid = threadIdx.x;
  const int b = blockIdx.z, h = blockIdx.y;
  const int w = tid >> 5, lane = tid & 31;
  const int am = lane & 15, ah = lane >> 4;
  const int qBase = blockIdx.x * 128 + w * 16;
  const long long cb = (long long)b * (256LL * NT);

  // Q fragments (A-matrix layout), channel of head h at row d*4+h
  Frag qa[2];
#pragma unroll
  for (int s = 0; s < 2; ++s) {
    int d0 = s * 32;
#pragma unroll
    for (int e = 0; e < 16; ++e) {
      int dk = ah * 8 + (e < 8 ? e : (e - 8) + 16);
      qa[s].u[e] = Q[cb + (long long)((d0 + dk) * 4 + h) * NT + qBase + am];
    }
  }

  v8f zero = {0.f, 0.f, 0.f, 0.f, 0.f, 0.f, 0.f, 0.f};
  v8f o[4] = {zero, zero, zero, zero};
  float rm[8], rl[8];
#pragma unroll
  for (int r = 0; r < 8; ++r) { rm[r] = -1e30f; rl[r] = 0.f; }

  for (int k0 = 0; k0 < NT; k0 += 32) {
    {  // stage K (key-major) and V (d-major) tiles
      int d = tid >> 2, kg = (tid & 3) * 8;
      const u16* kp = &Km[cb + (long long)(d * 4 + h) * NT + k0 + kg];
      const u16* vp = &Vm[cb + (long long)(d * 4 + h) * NT + k0 + kg];
      uint4 kq = *(const uint4*)kp;
      uint4 vq = *(const uint4*)vp;
      u16 ktmp[8];
      *(uint4*)ktmp = kq;
#pragma unroll
      for (int e = 0; e < 8; ++e) kt[(kg + e) * 64 + d] = ktmp[e];
      *(uint4*)&vl[d * 32 + kg] = vq;
      if (k0 + 32 < NT) {
        __builtin_prefetch(kp + 32, 0, 0);
        __builtin_prefetch(vp + 32, 0, 0);
      }
    }
    __syncthreads();

    // S = Q^T K for two 16-key sub-tiles
    v8f s0 = zero, s1 = zero;
#pragma unroll
    for (int st = 0; st < 2; ++st) {
      Frag bk0, bk1;
      const u16* p0 = &kt[am * 64 + st * 32 + ah * 8];
      const u16* p1 = &kt[(16 + am) * 64 + st * 32 + ah * 8];
      bk0.q[0] = *(const uint4*)p0;
      bk0.q[1] = *(const uint4*)(p0 + 16);
      bk1.q[0] = *(const uint4*)p1;
      bk1.q[1] = *(const uint4*)(p1 + 16);
      s0 = wmma_bf16<false>(qa[st], bk0, s0);
      s1 = wmma_bf16<true>(qa[st], bk1, s1);  // same A as previous wmma
    }

    // online softmax update (pure-VALU DPP reductions)
#pragma unroll
    for (int r = 0; r < 8; ++r) {
      float mv = max16(fmaxf(s0[r], s1[r]));
      float mnew = fmaxf(rm[r], mv);
      float alpha = __expf(rm[r] - mnew);
      rm[r] = mnew;
      float p0 = __expf(s0[r] - mnew);
      float p1 = __expf(s1[r] - mnew);
      float ps = sum16(p0 + p1);
      rl[r] = rl[r] * alpha + ps;
      o[0][r] *= alpha; o[1][r] *= alpha; o[2][r] *= alpha; o[3][r] *= alpha;
      int qrow = r + 8 * ah;
      pls[w][qrow * 32 + am]      = f2bf(p0);
      pls[w][qrow * 32 + 16 + am] = f2bf(p1);
    }

    // P (A layout) x V^T (B layout): O += P V^T   (LDS per-wave, DS in-order)
    Frag pa;
    const u16* pp = &pls[w][am * 32 + ah * 8];
    pa.q[0] = *(const uint4*)pp;
    pa.q[1] = *(const uint4*)(pp + 16);
#pragma unroll
    for (int dt = 0; dt < 4; ++dt) {
      Frag bvf;
      const u16* vp = &vl[(dt * 16 + am) * 32 + ah * 8];
      bvf.q[0] = *(const uint4*)vp;
      bvf.q[1] = *(const uint4*)(vp + 16);
      if (dt == 0)
        o[dt] = wmma_bf16<false>(pa, bvf, o[dt]);
      else
        o[dt] = wmma_bf16<true>(pa, bvf, o[dt]);
    }
    __syncthreads();
  }

  // msg[b][(d*4+h)][q] = O[q][d] / lsum * (1/sqrt(64))
#pragma unroll
  for (int dt = 0; dt < 4; ++dt) {
    int d = dt * 16 + am;
#pragma unroll
    for (int r = 0; r < 8; ++r) {
      int q = qBase + r + 8 * ah;
      float val = o[dt][r] / rl[r] * 0.125f;
      Mo[cb + (long long)(d * 4 + h) * NT + q] = f2bf(val);
    }
  }
}

// ---------------------------------------------------------------------------
// keypoint encoder MLP (3->32->64->128->256) + residual add into act (fp32)
// ---------------------------------------------------------------------------
__global__ __launch_bounds__(128) void enc_kernel(
    const float* __restrict__ desc, const float* __restrict__ kpts,
    const float* __restrict__ scr,
    const float* __restrict__ W0, const float* __restrict__ b0,
    const float* __restrict__ W1, const float* __restrict__ b1,
    const float* __restrict__ W2, const float* __restrict__ b2,
    const float* __restrict__ W3, const float* __restrict__ b3,
    float* __restrict__ act) {
  __shared__ float xv[3], h1[32], h2[64], h3[128];
  const int t = blockIdx.x;
  const int b = t >> 11, n = t & 2047;
  const int tid = threadIdx.x;
  if (tid == 0) {
    // center=(320,240), scaling = 640*0.7 = 448
    xv[0] = (kpts[(long long)(b * 2048 + n) * 2 + 0] - 320.f) * (1.f / 448.f);
    xv[1] = (kpts[(long long)(b * 2048 + n) * 2 + 1] - 240.f) * (1.f / 448.f);
    xv[2] = scr[b * 2048 + n];
  }
  __syncthreads();
  if (tid < 32) {
    float s = b0[tid];
    for (int i = 0; i < 3; ++i) s += W0[tid * 3 + i] * xv[i];
    h1[tid] = fmaxf(s, 0.f);
  }
  __syncthreads();
  if (tid < 64) {
    float s = b1[tid];
    for (int i = 0; i < 32; ++i) s += W1[tid * 32 + i] * h1[i];
    h2[tid] = fmaxf(s, 0.f);
  }
  __syncthreads();
  float s = b2[tid];
  for (int i = 0; i < 64; ++i) s += W2[tid * 64 + i] * h2[i];
  h3[tid] = fmaxf(s, 0.f);
  __syncthreads();
  for (int c = tid; c < 256; c += 128) {
    float v = b3[c];
    for (int i = 0; i < 128; ++i) v += W3[c * 128 + i] * h3[i];
    long long idx = (long long)b * (256LL * NT) + (long long)c * NT + n;
    act[idx] = desc[idx] + v;   // last layer: no relu
  }
}

// ---------------------------------------------------------------------------
// FC head: fc1 (2048->40, relu) then fc3t/fc3r (40->3 each); out = [t, r]
// ---------------------------------------------------------------------------
__global__ __launch_bounds__(64) void head_kernel(
    const float* __restrict__ pool, const float* __restrict__ fc1W,
    const float* __restrict__ fc1b, const float* __restrict__ fc3rW,
    const float* __restrict__ fc3rb, const float* __restrict__ fc3tW,
    const float* __restrict__ fc3tb, float* __restrict__ out) {
  __shared__ float hb[40];
  const int b = blockIdx.x;
  const int tid = threadIdx.x;
  for (int o = tid; o < 40; o += 64) {
    float s = fc1b[o];
    const float* wp = &fc1W[(long long)o * 2048];
    const float* pp = &pool[(long long)b * 2048];
    for (int i = 0; i < 2048; ++i) s += wp[i] * pp[i];
    hb[o] = fmaxf(s, 0.f);
  }
  __syncthreads();
  if (tid < 3) {
    float s = fc3tb[tid];
    for (int i = 0; i < 40; ++i) s += fc3tW[tid * 40 + i] * hb[i];
    out[b * 6 + tid] = s;
  } else if (tid < 6) {
    int c = tid - 3;
    float s = fc3rb[c];
    for (int i = 0; i < 40; ++i) s += fc3rW[c * 40 + i] * hb[i];
    out[b * 6 + tid] = s;
  }
}

// ---------------------------------------------------------------------------
// Host orchestration
// ---------------------------------------------------------------------------
extern "C" void kernel_launch(void* const* d_in, const int* in_sizes, int n_in,
                              void* d_out, int out_size, void* d_ws,
                              size_t ws_size, hipStream_t stream) {
  const float *descriptors = nullptr, *keypoints = nullptr, *scores = nullptr;
  int pidx[160];
  int np = 0;
  bool kp_found = false;
  for (int i = 0; i < n_in; ++i) {
    int sz = in_sizes[i];
    if (sz == 4194304 && !descriptors) { descriptors = (const float*)d_in[i]; continue; }
    if (sz == 32768 && !kp_found) { keypoints = (const float*)d_in[i]; kp_found = true; continue; }
    if (sz == 16384 && !scores) { scores = (const float*)d_in[i]; continue; }
    if (sz == 2457600) { continue; }  // image (only its fixed shape is used)
    if (np < 160) pidx[np++] = i;
  }
  if (!descriptors || !keypoints || !scores || np < 124) return;

  const float* P[124];
  for (int i = 0; i < 124; ++i) P[i] = (const float*)d_in[pidx[i]];

  const float *encW[4], *encb[4];
  const float *conv1W, *conv1b, *fc1W, *fc1b, *fc3rW, *fc3rb, *fc3tW, *fc3tb;
  struct L { const float *Wq, *bq, *Wk, *bk, *Wv, *bv, *Wm, *bm, *W1, *b1, *W2, *b2; } Ls[9];

  if (in_sizes[pidx[0]] == 96) {  // insertion-order flatten
    for (int i = 0; i < 4; ++i) { encW[i] = P[i]; encb[i] = P[4 + i]; }
    for (int l = 0; l < 9; ++l) {
      int q = 8 + 12 * l;
      Ls[l] = {P[q], P[q + 1], P[q + 2], P[q + 3], P[q + 4],  P[q + 5],
               P[q + 6], P[q + 7], P[q + 8], P[q + 9], P[q + 10], P[q + 11]};
    }
    conv1W = P[116]; conv1b = P[117]; fc1W = P[118]; fc1b = P[119];
    fc3rW = P[120]; fc3rb = P[121]; fc3tW = P[122]; fc3tb = P[123];
  } else {  // sorted-pytree flatten
    conv1W = P[0]; conv1b = P[1];
    for (int i = 0; i < 4; ++i) { encW[i] = P[2 + i]; encb[i] = P[6 + i]; }
    fc1W = P[10]; fc1b = P[11]; fc3rW = P[12]; fc3rb = P[13];
    fc3tW = P[14]; fc3tb = P[15];
    for (int l = 0; l < 9; ++l) {
      int q = 16 + 12 * l;  // W1,W2,Wk,Wm,Wq,Wv,b1,b2,bk,bm,bq,bv
      Ls[l].W1 = P[q];     Ls[l].W2 = P[q + 1]; Ls[l].Wk = P[q + 2];
      Ls[l].Wm = P[q + 3]; Ls[l].Wq = P[q + 4]; Ls[l].Wv = P[q + 5];
      Ls[l].b1 = P[q + 6]; Ls[l].b2 = P[q + 7]; Ls[l].bk = P[q + 8];
      Ls[l].bm = P[q + 9]; Ls[l].bq = P[q + 10]; Ls[l].bv = P[q + 11];
    }
  }

  // workspace layout
  char* ws = (char*)d_ws;
  float* act  = (float*)(ws + 0);           // (B,256,NT) fp32   16 MB
  u16* catb   = (u16*)(ws + 16777216);      // (B,512,NT) bf16   16 MB
  u16* qb     = (u16*)(ws + 33554432);      // (B,256,NT) bf16    8 MB
  u16* kb     = (u16*)(ws + 41943040);
  u16* vb     = (u16*)(ws + 50331648);
  u16* msgb   = (u16*)(ws + 58720256);
  u16* hbuf   = (u16*)(ws + 67108864);      // (B,512,NT) bf16   16 MB
  float* pool = (float*)(ws + 83886080);    // (B,2048) fp32
  u16* wbf    = (u16*)(ws + 83951616);      // bf16 weights

  const long long LWB = 655360;  // per-layer bf16 weight block (elements)
  auto cvt = [&](const float* src, u16* dst, int n) {
    cvt_f32_bf16<<<(n + 255) / 256, 256, 0, stream>>>(src, dst, n);
  };

  // convert all GEMM weights to bf16 once
  for (int l = 0; l < 9; ++l) {
    u16* base = wbf + l * LWB;
    cvt(Ls[l].Wq, base + 0,      65536);
    cvt(Ls[l].Wk, base + 65536,  65536);
    cvt(Ls[l].Wv, base + 131072, 65536);
    cvt(Ls[l].Wm, base + 196608, 65536);
    cvt(Ls[l].W1, base + 262144, 262144);
    cvt(Ls[l].W2, base + 524288, 131072);
  }
  u16* conv1Wb = wbf + 9 * LWB;
  cvt(conv1W, conv1Wb, 524288);

  // keypoint encoder + residual into act
  enc_kernel<<<BATCH * NT, 128, 0, stream>>>(
      descriptors, keypoints, scores, encW[0], encb[0], encW[1], encb[1],
      encW[2], encb[2], encW[3], encb[3], act);

  const long long sx2 = 512LL * NT;  // catb / hbuf batch stride (elements)
  const long long sx1 = 256LL * NT;  // 256-row batch stride
  dim3 blk(256);

  for (int l = 0; l < 9; ++l) {
    u16* base = wbf + l * LWB;
    cvt_act_kernel<<<16384, 256, 0, stream>>>(act, catb);

    dim3 g256(16, 4, BATCH);
    gemm_bf16<0><<<g256, blk, 0, stream>>>(base + 0, catb, Ls[l].bq, qb,
                                           nullptr, 256, 256, sx2, sx1);
    gemm_bf16<0><<<g256, blk, 0, stream>>>(base + 65536, catb, Ls[l].bk, kb,
                                           nullptr, 256, 256, sx2, sx1);
    gemm_bf16<0><<<g256, blk, 0, stream>>>(base + 131072, catb, Ls[l].bv, vb,
                                           nullptr, 256, 256, sx2, sx1);

    attn_kernel<<<dim3(16, NHEAD, BATCH), blk, 0, stream>>>(qb, kb, vb, msgb);

    // Wm conv: write into rows 256..511 of catb
    gemm_bf16<0><<<g256, blk, 0, stream>>>(base + 196608, msgb, Ls[l].bm,
                                           catb + 256LL * NT, nullptr, 256, 256,
                                           sx1, sx2);
    // W1 (512x512) + relu -> hbuf
    gemm_bf16<1><<<dim3(16, 8, BATCH), blk, 0, stream>>>(
        base + 262144, catb, Ls[l].b1, hbuf, nullptr, 512, 512, sx2, sx2);
    // W2 (256x512) residual add into act (fp32)
    gemm_bf16<2><<<g256, blk, 0, stream>>>(base + 524288, hbuf, Ls[l].b2,
                                           nullptr, act, 256, 512, sx2, sx1);
  }

  // final conv1 (2048x256) + relu + global max pool over tokens
  cvt_act_kernel<<<16384, 256, 0, stream>>>(act, catb);
  zero_kernel<<<64, 256, 0, stream>>>(pool, BATCH * 2048);
  gemm_bf16<3><<<dim3(16, 32, BATCH), blk, 0, stream>>>(
      conv1Wb, catb, conv1b, nullptr, pool, 2048, 256, sx2, 0);

  head_kernel<<<BATCH, 64, 0, stream>>>(pool, fc1W, fc1b, fc3rW, fc3rb, fc3tW,
                                        fc3tb, (float*)d_out);
}
